// ExpertsFeedForward_74380243632812
// MI455X (gfx1250) — compile-verified, matrix operand order checked by reading, not run on hardware
//
#include <hip/hip_runtime.h>
#include <hip/hip_bf16.h>
#include <math.h>

#define NTOK 8192
#define DM   1024
#define HD   4096
#define NE   8
#define TM   32
#define MAXTILES 520   // sum ceil(c_e/32) <= 16384/32 + 8

typedef __attribute__((ext_vector_type(16))) __bf16 v16bf;
typedef __attribute__((ext_vector_type(4)))  __bf16 v4bf;
typedef __attribute__((ext_vector_type(8)))  float  v8f;

// LDS fragment address for 16-bit WMMA A/B fragments (ISA 7.12.2):
// lane = row + 16*((k>>3)&1), elem = (k&7) + 8*((k>>4)&1)
__device__ __forceinline__ int frag_off(int tile, int row, int k) {
  int l = row + (((k >> 3) & 1) << 4);
  int j = (k & 7) + (((k >> 4) & 1) << 3);
  return (tile * 32 + l) * 16 + j;
}

// gelu(x) = 0.5x(1+tanh(u)) = x*sigmoid(2u); branch-free: v_exp_f32 + v_rcp_f32
__device__ __forceinline__ float gelu_tanh(float x) {
  float t = 1.5957691216057308f * (x + 0.044715f * x * x * x);
  return x * __builtin_amdgcn_rcpf(1.0f + __expf(-t));
}

__global__ void moe_init(unsigned* p) {
  if (threadIdx.x < 17) p[threadIdx.x] = 0u;   // usage[8], zacc[1], cursor[8]
}

__global__ void moe_router(const float* __restrict__ X, const float* __restrict__ G,
                           float* __restrict__ usage, float* __restrict__ zacc,
                           int* __restrict__ cursor, int* __restrict__ tok_list,
                           float* __restrict__ tok_w) {
  const int w = threadIdx.x >> 5;
  const int l = threadIdx.x & 31;
  const int n = blockIdx.x * 8 + w;
  const float* xr = X + (size_t)n * DM;
  float acc[NE];
#pragma unroll
  for (int e = 0; e < NE; ++e) acc[e] = 0.0f;
  for (int kk = 0; kk < DM / 32; ++kk) {
    int k = kk * 32 + l;
    float xv = xr[k];
    const float4* g4 = (const float4*)(G + k * NE);
    float4 ga = g4[0], gb = g4[1];
    acc[0] += xv * ga.x; acc[1] += xv * ga.y;
    acc[2] += xv * ga.z; acc[3] += xv * ga.w;
    acc[4] += xv * gb.x; acc[5] += xv * gb.y;
    acc[6] += xv * gb.z; acc[7] += xv * gb.w;
  }
#pragma unroll
  for (int e = 0; e < NE; ++e)
    for (int off = 16; off > 0; off >>= 1)
      acc[e] += __shfl_xor(acc[e], off, 32);
  if (l == 0) {
    int a1 = 0; float s1 = acc[0];
#pragma unroll
    for (int e = 1; e < NE; ++e) if (acc[e] > s1) { s1 = acc[e]; a1 = e; }
    int a2 = (a1 == 0) ? 1 : 0; float s2 = acc[a2];
#pragma unroll
    for (int e = 0; e < NE; ++e) if (e != a1 && acc[e] > s2) { s2 = acc[e]; a2 = e; }
    float e2 = __expf(s2 - s1);
    float rs = __builtin_amdgcn_rcpf(1.0f + e2);
    float w1 = rs;
    float w2 = e2 * rs;
    float sum = 0.0f;
#pragma unroll
    for (int e = 0; e < NE; ++e) sum += __expf(acc[e] - s1);
    float lse = s1 + __logf(sum);
    atomicAdd(zacc, lse * lse);
    atomicAdd(usage + a1, w1);
    atomicAdd(usage + a2, w2);
    int p1 = atomicAdd(cursor + a1, 1);
    tok_list[a1 * NTOK + p1] = n; tok_w[a1 * NTOK + p1] = w1;
    int p2 = atomicAdd(cursor + a2, 1);
    tok_list[a2 * NTOK + p2] = n; tok_w[a2 * NTOK + p2] = w2;
  }
}

__global__ void moe_setup_tiles(const int* __restrict__ cursor, int* __restrict__ tile_e,
                                int* __restrict__ tile_r0, int* __restrict__ ntile) {
  int t = 0;
  for (int e = 0; e < NE; ++e) {
    int c = cursor[e];
    for (int r0 = 0; r0 < c; r0 += TM) { tile_e[t] = e; tile_r0[t] = r0; ++t; }
  }
  *ntile = t;
}

__global__ void moe_finalize(const float* __restrict__ usage, const float* __restrict__ zacc,
                             float* __restrict__ out_scalar) {
  float s = 0.f;
  for (int e = 0; e < NE; ++e) s += usage[e];
  float mean = s / (float)NE;
  float var = 0.f;
  for (int e = 0; e < NE; ++e) { float d = usage[e] - mean; var += d * d; }
  var /= (float)NE;
  float bal = sqrtf(var) / mean * 0.01f;
  float z = (*zacc / (float)NTOK) * 0.001f;
  *out_scalar = bal + z;
}

// Fused FFN, 32-token tile: out[32,1024] = gelu(X[32,1024]@W1 + b1)[:,chunk] @ W2[chunk,:] (+b2)
// 8 waves; wave w owns output N-tiles w*8..w*8+7 (both M halves) and hidden cols w*16..w*16+15.
// W1/W2 staging buffers are LDS double-buffered: one barrier per k-slice, so the next
// slice's global loads overlap the current slice's WMMAs.
__global__ __launch_bounds__(256)
void moe_ffn(const float* __restrict__ X,
             const float* __restrict__ W1base, const float* __restrict__ B1base,
             const float* __restrict__ W2base, const float* __restrict__ B2base,
             float* __restrict__ OUT, int mode,
             const int* __restrict__ tile_e, const int* __restrict__ tile_r0,
             const int* __restrict__ ntile, const int* __restrict__ cursor,
             const int* __restrict__ tok_list, const float* __restrict__ tok_w) {
  extern __shared__ char smem_raw[];
  __bf16* Afrag  = (__bf16*)smem_raw;          // 64 tiles (32 ks x 2 m2) = 32768 halves
  __bf16* Bfrag1 = Afrag + 32768;              // 2 x 32 tiles = 32768 halves (double buffer)
  __bf16* Hfrag  = Bfrag1 + 32768;             // 8 tiles (4 ks x 2 m2) = 4096 halves
  __bf16* Bfrag2 = Hfrag + 4096;               // 2 x 64 tiles = 65536 halves (double buffer)
  __shared__ int   s_tok[TM];
  __shared__ float s_wt[TM];

  const int w  = threadIdx.x >> 5;
  const int l  = threadIdx.x & 31;
  const int ln = l & 15;
  const int lh = l >> 4;

  const float *W1, *B1, *W2, *B2;
  int e = 0, r0 = 0, cnt = 0, tb = 0;
  if (mode) {
    if ((int)blockIdx.x >= *ntile) return;
    e   = tile_e[blockIdx.x];
    r0  = tile_r0[blockIdx.x];
    cnt = cursor[e];
    W1 = W1base + (size_t)e * DM * HD;  B1 = B1base + (size_t)e * HD;
    W2 = W2base + (size_t)e * HD * DM;  B2 = B2base + (size_t)e * DM;
  } else {
    tb = blockIdx.x * TM;
    W1 = W1base; B1 = B1base; W2 = W2base; B2 = B2base;
  }

  if (threadIdx.x < TM) {
    int gtok; float wt = 1.0f;
    if (mode) {
      int rr = r0 + threadIdx.x;
      if (rr < cnt) { gtok = tok_list[e * NTOK + rr]; wt = tok_w[e * NTOK + rr]; }
      else gtok = -1;
    } else {
      gtok = tb + threadIdx.x;
    }
    s_tok[threadIdx.x] = gtok;
    s_wt[threadIdx.x]  = wt;
  }
  __syncthreads();

  // Stage X tile (32 x 1024 fp32 -> bf16) into A-fragment layout; packed b64 LDS stores.
#pragma unroll 4
  for (int it = 0; it < 32; ++it) {
    int i   = threadIdx.x + 256 * it;     // 0..8191 float4 slots
    int row = i >> 8;                     // 0..31
    int k0  = (i & 255) << 2;             // 0..1020, step 4
    int tk  = s_tok[row];
    float4 v = (tk >= 0) ? *(const float4*)(X + (size_t)tk * DM + k0)
                         : make_float4(0.f, 0.f, 0.f, 0.f);
    v4bf hv = { (__bf16)v.x, (__bf16)v.y, (__bf16)v.z, (__bf16)v.w };
    *(v4bf*)(Afrag + frag_off(((k0 >> 5) << 1) + (row >> 4), row & 15, k0 & 31)) = hv;
  }
  __syncthreads();

  v8f acc[16];
#pragma unroll
  for (int i = 0; i < 16; ++i) acc[i] = (v8f){0.f,0.f,0.f,0.f,0.f,0.f,0.f,0.f};

  for (int h = 0; h < HD / 128; ++h) {
    const int hc0 = h * 128;
    // ---------- phase A: hidden[32,128] = X @ W1[:, hc0:hc0+128] ----------
    v8f hacc[2];
    hacc[0] = (v8f){0.f,0.f,0.f,0.f,0.f,0.f,0.f,0.f};
    hacc[1] = (v8f){0.f,0.f,0.f,0.f,0.f,0.f,0.f,0.f};
    for (int kb = 0; kb < 8; ++kb) {              // K over D in blocks of 128
      __bf16* buf1 = Bfrag1 + ((kb & 1) << 14);   // 16384-half double buffer
      if (kb < 7) {                               // pull next slice toward L2
        int pr = threadIdx.x >> 1;
        int pc = (threadIdx.x & 1) << 6;
        __builtin_prefetch(W1 + (size_t)(((kb + 1) << 7) + pr) * HD + hc0 + pc, 0, 1);
      }
#pragma unroll 4
      for (int it = 0; it < 16; ++it) {           // stage W1 [128 x 128] -> B fragments
        int i  = threadIdx.x + 256 * it;          // 0..4095 float4 slots
        int kr = i >> 5;                          // 0..127
        int c  = (i & 31) << 2;                   // 0..124, step 4
        float4 v = *(const float4*)(W1 + (size_t)((kb << 7) + kr) * HD + hc0 + c);
        int base = frag_off(((kr >> 5) << 3) + (c >> 4), c & 15, kr & 31);
        buf1[base]      = (__bf16)v.x;            // consecutive n -> lane+1 (stride 16)
        buf1[base + 16] = (__bf16)v.y;
        buf1[base + 32] = (__bf16)v.z;
        buf1[base + 48] = (__bf16)v.w;
      }
      __syncthreads();                            // single barrier: dbl-buffer covers WAR
#pragma unroll
      for (int ksl = 0; ksl < 4; ++ksl) {
        int ks = (kb << 2) + ksl;
        v16bf b = *(const v16bf*)(buf1 + ((((ksl << 3) + w) * 32 + l) << 4));
#pragma unroll
        for (int m2 = 0; m2 < 2; ++m2) {
          v16bf a = *(const v16bf*)(Afrag + ((((ks << 1) + m2) * 32 + l) << 4));
          hacc[m2] = __builtin_amdgcn_wmma_f32_16x16x32_bf16(false, a, false, b,
                                                             (short)0, hacc[m2], false, false);
        }
      }
    }
    __syncthreads();                              // all phase-A reads done before Hfrag write
    // bias + gelu, write hidden chunk to LDS as phase-B A fragments (bf16)
    {
      int hidcol = (w << 4) + ln;
      float b1v = B1[hc0 + hidcol];
#pragma unroll
      for (int m2 = 0; m2 < 2; ++m2) {
#pragma unroll
        for (int r = 0; r < 8; ++r) {
          float hv = gelu_tanh(hacc[m2][r] + b1v);
          Hfrag[frag_off(((hidcol >> 5) << 1) + m2, r + (lh << 3), hidcol & 31)] = (__bf16)hv;
        }
      }
    }
    __syncthreads();
    // ---------- phase B: out[32,1024] += hidden @ W2[hc0:hc0+128, :] ----------
    for (int ks2 = 0; ks2 < 4; ++ks2) {
      __bf16* buf2 = Bfrag2 + ((ks2 & 1) << 15);  // 32768-half double buffer
      if (ks2 < 3) {                              // pull next slice toward L2
        int pr = threadIdx.x >> 3;
        int pc = (threadIdx.x & 7) << 7;
        __builtin_prefetch(W2 + (size_t)(hc0 + ((ks2 + 1) << 5) + pr) * DM + pc, 0, 1);
      }
#pragma unroll 4
      for (int it = 0; it < 32; ++it) {           // stage W2 [32 x 1024] -> B fragments
        int i  = threadIdx.x + 256 * it;          // 0..8191 float4 slots
        int kr = i >> 8;                          // 0..31
        int c  = (i & 255) << 2;                  // 0..1020, step 4
        float4 v = *(const float4*)(W2 + (size_t)(hc0 + (ks2 << 5) + kr) * DM + c);
        int base = frag_off(c >> 4, c & 15, kr);
        buf2[base]      = (__bf16)v.x;
        buf2[base + 16] = (__bf16)v.y;
        buf2[base + 32] = (__bf16)v.z;
        buf2[base + 48] = (__bf16)v.w;
      }
      __syncthreads();                            // single barrier: dbl-buffer covers WAR
      v16bf ha0 = *(const v16bf*)(Hfrag + ((((ks2 << 1) + 0) * 32 + l) << 4));
      v16bf ha1 = *(const v16bf*)(Hfrag + ((((ks2 << 1) + 1) * 32 + l) << 4));
#pragma unroll
      for (int i2 = 0; i2 < 8; ++i2) {
        v16bf b = *(const v16bf*)(buf2 + ((((w << 3) + i2) * 32 + l) << 4));
        acc[i2]     = __builtin_amdgcn_wmma_f32_16x16x32_bf16(false, ha0, false, b,
                                                              (short)0, acc[i2], false, false);
        acc[8 + i2] = __builtin_amdgcn_wmma_f32_16x16x32_bf16(false, ha1, false, b,
                                                              (short)0, acc[8 + i2], false, false);
      }
    }
    __syncthreads();                              // phase-B reads done before next Hfrag write
  }

  // ---------- epilogue ----------
#pragma unroll
  for (int i2 = 0; i2 < 8; ++i2) {
    int nc = (((w << 3) + i2) << 4) + ln;
    float bias = B2[nc];
#pragma unroll
    for (int m2 = 0; m2 < 2; ++m2) {
#pragma unroll
      for (int r = 0; r < 8; ++r) {
        int m  = (m2 << 4) + r + (lh << 3);
        int tk = s_tok[m];
        if (tk < 0) continue;
        float v = acc[m2 * 8 + i2][r] + bias;
        if (mode) atomicAdd(&OUT[(size_t)tk * DM + nc], s_wt[m] * v);
        else      OUT[(size_t)tk * DM + nc] = v;
      }
    }
  }
}

extern "C" void kernel_launch(void* const* d_in, const int* in_sizes, int n_in,
                              void* d_out, int out_size, void* d_ws, size_t ws_size,
                              hipStream_t stream) {
  (void)in_sizes; (void)n_in; (void)out_size; (void)ws_size;
  const float* x    = (const float*)d_in[0];
  const float* gate = (const float*)d_in[1];
  const float* kw   = (const float*)d_in[2];
  const float* kb   = (const float*)d_in[3];
  const float* vw   = (const float*)d_in[4];
  const float* vb   = (const float*)d_in[5];
  const float* skw  = (const float*)d_in[6];
  const float* skb  = (const float*)d_in[7];
  const float* svw  = (const float*)d_in[8];
  const float* svb  = (const float*)d_in[9];
  float* out = (float*)d_out;

  float* usage    = (float*)d_ws;                 // 8
  float* zacc     = usage + 8;                    // 1
  int*   cursor   = (int*)(zacc + 1);             // 8
  int*   ntile    = cursor + 8;                   // 1
  int*   tok_list = ntile + 1;                    // NE*NTOK
  float* tok_w    = (float*)(tok_list + NE * NTOK);
  int*   tile_e   = (int*)(tok_w + NE * NTOK);    // 528
  int*   tile_r0  = tile_e + 528;                 // 528

  moe_init<<<1, 32, 0, stream>>>((unsigned*)usage);
  moe_router<<<NTOK / 8, 256, 0, stream>>>(x, gate, usage, zacc, cursor, tok_list, tok_w);
  moe_setup_tiles<<<1, 1, 0, stream>>>(cursor, tile_e, tile_r0, ntile);
  moe_finalize<<<1, 1, 0, stream>>>(usage, zacc, out + (size_t)NTOK * DM);

  const size_t smem = (size_t)(32768 + 32768 + 4096 + 65536) * 2u;  // 264 KB LDS
  // shared expert: initializes OUT
  moe_ffn<<<NTOK / TM, 256, smem, stream>>>(x, skw, skb, svw, svb, out, 0,
                                            tile_e, tile_r0, ntile, cursor, tok_list, tok_w);
  // routed experts: weighted atomic combine
  moe_ffn<<<MAXTILES, 256, smem, stream>>>(x, kw, kb, vw, vb, out, 1,
                                           tile_e, tile_r0, ntile, cursor, tok_list, tok_w);
}
